// NSP_59382217835209
// MI455X (gfx1250) — matrix operand order repaired and checked by reading, not compiled, not run on hardware
//
#include <hip/hip_runtime.h>
#include <hip/hip_bf16.h>
#include <math.h>

// CDNA5 (gfx1250) wave32 WMMA types
typedef __attribute__((ext_vector_type(16))) _Float16 v16h;
typedef __attribute__((ext_vector_type(8)))  float    v8f;

#define K_SCOPE 25.0f
#define K_ENV   100.0f
#define WIN     27
#define KCELLS  729           // 27*27
#define NCHUNK  23            // ceil(729/32)
#define MAPD    4096

__device__ __forceinline__ void add_force_label(
    float cnt, float sr, float sc, float w, float p1,
    bool deg_r, bool deg_c, bool both, bool rlt, bool clt,
    float& Fr, float& Fc)
{
    float denom  = fmaxf(cnt, 1.0f);
    float mean_r = sr / denom;
    float mean_c = sc / denom;
    float fr = 0.0f, fc = 0.0f;
    if (deg_r && !deg_c) {
        float disA = clt ? mean_c : (K_SCOPE - mean_c);
        if (disA != 0.0f) {
            float mag = (w * K_ENV) / disA;
            fc = clt ? -mag : mag;
        }
    } else if (!deg_r && deg_c) {
        float disB = rlt ? (mean_r + p1) : (K_SCOPE - mean_r);
        if (disB != 0.0f) {
            float mag = (w * K_ENV) / disB;
            fr = rlt ? -mag : mag;
        }
    } else {
        // FC branch (both-degenerate case falls here but is zeroed below)
        float cr  = rlt ? 0.0f : K_SCOPE;
        float cc  = clt ? 0.0f : K_SCOPE;
        float dR  = mean_r - cr;
        float dC  = mean_c - cc;
        float dis = sqrtf(dR * dR + dC * dC);
        if (dis != 0.0f) {
            float inv = -(w * K_ENV) / (dis * dis);
            fr = inv * dR;
            fc = inv * dC;
        }
    }
    if (!(cnt > 0.0f) || both) { fr = 0.0f; fc = 0.0f; }
    Fr += fr;
    Fc += fc;
}

// One wave handles 16 particles. Window reductions for labels {5,3,4} run as
// D = A x B + C on v_wmma_f32_16x16x32_f16:
//   A[m][k] = match(particle m, cell k) in {0,1}  (f16 exact)
//   B[k][{0,1,2}] = {1, row(k), col(k)}           (f16 exact, ints <= 27)
//   D[m][{0,1,2}] = {cnt, sum_r, sum_c}           (f32 accumulate, exact)
__global__ __launch_bounds__(256)
void NSP_forces_wmma_kernel(const float2* __restrict__ step,
                            const float2* __restrict__ ff,
                            const float2* __restrict__ vel,
                            const int*    __restrict__ map,
                            float2*       __restrict__ out,
                            int nPart)
{
    // Precomputed B fragments: particle-independent, shared by all waves.
    // btab[chunk][lane][h] in contiguous order: K = 16*(lane/16) + h.
    __shared__ __align__(32) _Float16 btab[NCHUNK][32][16];
    // [wave][label][vgpr][lane] accumulator bounce for cross-lane extraction
    __shared__ float red[8][3][8][32];

    const int tid   = threadIdx.x;
    const int lane  = tid & 31;
    const int wib   = tid >> 5;
    const int gwave = (blockIdx.x * blockDim.x + tid) >> 5;
    const int mrow  = lane & 15;                 // particle row M for this lane

    // ---- fill B table once per block (branch-free: fmaf column weights) ----
    for (int idx = tid; idx < NCHUNK * 32; idx += 256) {
        const int chunk = idx >> 5;
        const int ln    = idx & 31;
        const int bbase = (ln & 16) ? 16 : 0;    // B: K = bbase + h
        const int bcol  = ln & 15;               // B column N carried by lane
        const float w0  = (bcol == 0) ? 1.0f : 0.0f;
        const float w1  = (bcol == 1) ? 1.0f : 0.0f;
        const float w2  = (bcol == 2) ? 1.0f : 0.0f;
        #pragma unroll
        for (int h = 0; h < 16; ++h) {
            const unsigned k  = (unsigned)(chunk * 32 + bbase + h);
            const unsigned kr = k / 27u;
            const unsigned kc = k - kr * 27u;
            const float bv = fmaf(w1, (float)kr, fmaf(w2, (float)kc, w0));
            btab[chunk][ln][h] = (_Float16)bv;
        }
    }

    int p = gwave * 16 + mrow;
    if (p >= nPart) p = nPart - 1;               // N divides evenly; safety clamp

    // ---- per-particle window setup (lanes L, L+16 compute redundantly) ----
    const float2 st = step[p];
    const float2 f0 = ff[p];
    const float2 vl = vel[p];
    const float ori_r = st.x + f0.x;
    const float ori_c = st.y + f0.y;
    const float sgr = (vl.x > 0.0f) ? 1.0f : ((vl.x < 0.0f) ? -1.0f : 0.0f);
    const float sgc = (vl.y > 0.0f) ? 1.0f : ((vl.y < 0.0f) ? -1.0f : 0.0f);
    const int r0 = (int)floorf(ori_r);
    const int c0 = (int)floorf(ori_c);
    const int r1 = (int)floorf(ori_r + sgr * K_SCOPE);
    const int c1 = (int)floorf(ori_c + sgc * K_SCOPE);
    const int rmin = min(r0, r1), rmax = max(r0, r1);
    const int cmin = min(c0, c1), cmax = max(c0, c1);
    const bool deg_r = (r0 == r1);
    const bool deg_c = (c0 == c1);
    const bool both  = deg_r && deg_c;
    const int dr = rmax - rmin;
    const int dc = cmax - cmin;

    // ---- closed-form row/col validity bitmasks (27 bits each) ----
    // mr bits: lr < dr  -> (1<<dr)-1, degenerate (dr==0) -> bit 0
    // rin bits: i in [max(0,-rmin), min(27, 4096-rmin))
    // Bits >= 27 stay zero, which also kills padded k >= 729 (kr == 27).
    unsigned rowmask, colmask;
    {
        const unsigned mrb = dr ? ((1u << dr) - 1u) : 1u;
        const unsigned mcb = dc ? ((1u << dc) - 1u) : 1u;
        const int rlo = min(max(-rmin, 0), WIN);
        const int rhi = min(max(MAPD - rmin, 0), WIN);
        const int clo = min(max(-cmin, 0), WIN);
        const int chi = min(max(MAPD - cmin, 0), WIN);
        const unsigned rinb = ((1u << rhi) - 1u) & ~((1u << rlo) - 1u);
        const unsigned cinb = ((1u << chi) - 1u) & ~((1u << clo) - 1u);
        rowmask = mrb & rinb;
        colmask = mcb & cinb;
        if (both) rowmask = 0u;                  // ~both_deg term of the mask
    }

    // ---- WMMA A-fragment slot -> K mapping (contiguous order, same as B table) ----
    const int base0 = (lane < 16) ? 0 : 16;      // K = base0+{0..15}

    v8f acc5 = {0.f,0.f,0.f,0.f,0.f,0.f,0.f,0.f};
    v8f acc3 = {0.f,0.f,0.f,0.f,0.f,0.f,0.f,0.f};
    v8f acc4 = {0.f,0.f,0.f,0.f,0.f,0.f,0.f,0.f};

    __syncthreads();                             // B table ready

    for (int chunk = 0; chunk < NCHUNK; ++chunk) {
        const int k0 = chunk * 32;

        // ---- phase 0: fetch B fragment from LDS (2x ds_load_b128) ----
        const v16h bfr = *(const v16h*)btab[chunk][lane];

        // ---- phase 1: issue all 16 gathers (single wait, 16-deep MLP) ----
        int      wv[16];
        unsigned krA[16], kcA[16];
        #pragma unroll
        for (int h = 0; h < 16; ++h) {
            const unsigned k  = (unsigned)(k0 + base0 + h);
            const unsigned kr = k / 27u;
            const unsigned kc = k - kr * 27u;
            const int row = rmin + (int)kr;
            const int col = cmin + (int)kc;
            const int rcl = min(max(row, 0), MAPD - 1);
            const int ccl = min(max(col, 0), MAPD - 1);
            const unsigned off = ((unsigned)rcl << 12) | (unsigned)ccl;
            wv[h]  = map[off];                   // L2-resident gather
            krA[h] = kr;
            kcA[h] = kc;
        }

        // ---- phase 2: pack match bits into A fragments ----
        v16h a5, a3, a4;
        #pragma unroll
        for (int h = 0; h < 16; ++h) {
            const bool msk = (((rowmask >> krA[h]) & (colmask >> kcA[h])) & 1u) != 0u;
            const int  w   = wv[h];
            a5[h] = (msk & (w == 5)) ? (_Float16)1.0f : (_Float16)0.0f;
            a3[h] = (msk & (w == 3)) ? (_Float16)1.0f : (_Float16)0.0f;
            a4[h] = (msk & (w == 4)) ? (_Float16)1.0f : (_Float16)0.0f;
        }

        // ---- phase 3: accumulate {cnt, sum_r, sum_c} per label ----
        acc5 = __builtin_amdgcn_wmma_f32_16x16x32_f16(false, a5, false, bfr,
                                                      (short)0, acc5, false, false);
        acc3 = __builtin_amdgcn_wmma_f32_16x16x32_f16(false, a3, false, bfr,
                                                      (short)0, acc3, false, false);
        acc4 = __builtin_amdgcn_wmma_f32_16x16x32_f16(false, a4, false, bfr,
                                                      (short)0, acc4, false, false);
    }

    // ---- spill accumulators to LDS for cross-lane extraction ----
    #pragma unroll
    for (int v = 0; v < 8; ++v) {
        red[wib][0][v][lane] = acc5[v];
        red[wib][1][v][lane] = acc3[v];
        red[wib][2][v][lane] = acc4[v];
    }
    __syncthreads();

    // ---- epilogue: lanes 0..15, one particle each ----
    if (lane < 16) {
        // C/D layout: element (M,N) -> VGPR M%8, lane (M/8)*16 + N
        const int vi  = mrow & 7;
        const int off = (mrow >> 3) * 16;
        const float cnt5 = red[wib][0][vi][off + 0];
        const float sr5  = red[wib][0][vi][off + 1];
        const float sc5  = red[wib][0][vi][off + 2];
        const float cnt3 = red[wib][1][vi][off + 0];
        const float sr3  = red[wib][1][vi][off + 1];
        const float sc3  = red[wib][1][vi][off + 2];
        const float cnt4 = red[wib][2][vi][off + 0];
        const float sr4  = red[wib][2][vi][off + 1];
        const float sc4  = red[wib][2][vi][off + 2];

        const bool rlt = (r0 < r1);
        const bool clt = (c0 < c1);
        float Fr = 0.0f, Fc = 0.0f;
        add_force_label(cnt5, sr5, sc5, 1.0f, 1.0f, deg_r, deg_c, both, rlt, clt, Fr, Fc);
        add_force_label(cnt3, sr3, sc3, 1.0f, 0.0f, deg_r, deg_c, both, rlt, clt, Fr, Fc);
        add_force_label(cnt4, sr4, sc4, 1.5f, 0.0f, deg_r, deg_c, both, rlt, clt, Fr, Fc);

        const int pw = gwave * 16 + mrow;
        if (pw < nPart) {
            float2 F; F.x = Fr; F.y = Fc;
            out[pw] = F;
        }
    }
}

extern "C" void kernel_launch(void* const* d_in, const int* in_sizes, int n_in,
                              void* d_out, int out_size, void* d_ws, size_t ws_size,
                              hipStream_t stream) {
    (void)n_in; (void)out_size; (void)d_ws; (void)ws_size;
    const float2* step = (const float2*)d_in[0];   // current_step (N,2) f32
    const float2* ff   = (const float2*)d_in[1];   // first_frame  (N,2) f32
    const float2* vel  = (const float2*)d_in[2];   // current_vel  (N,2) f32
    const int*    map  = (const int*)   d_in[3];   // semantic_map (4096,4096) i32
    // d_in[4] = F0 (zeros) unused: F2 = zeros_like(F0) + forces
    float2* out = (float2*)d_out;                  // (N,2) f32

    const int nPart = in_sizes[0] / 2;             // 65536
    // 256 threads = 8 waves/block, 16 particles/wave -> 128 particles/block
    const int blocks = (nPart + 127) / 128;        // 512
    NSP_forces_wmma_kernel<<<blocks, 256, 0, stream>>>(step, ff, vel, map, out, nPart);
}
